// TrajectoryDecoder_13305808683229
// MI455X (gfx1250) — matrix operand order repaired and checked by reading, not compiled, not run on hardware
//
#include <hip/hip_runtime.h>
#include <hip/hip_bf16.h>

// Problem constants (from reference): B=16384, H=256, O=2, M=3, L=2, T=80
#define B_TOTAL 16384
#define HDIM    256
#define T_STEPS 80
#define R       32          // batch rows per workgroup (2 M-tiles of 16)
#define HP      264         // padded LDS row stride in bf16 elements (528B, bank-conflict-free)

typedef __attribute__((ext_vector_type(16))) __bf16 v16bf;
typedef __attribute__((ext_vector_type(8)))  float  v8f;

union BF16Frag {
  v16bf v;
  uint4 q[2];
  unsigned short s[16];
};

__device__ __forceinline__ unsigned short f2bf(float f) {
  unsigned int u = __float_as_uint(f);
  u += 0x7FFFu + ((u >> 16) & 1u);        // round-to-nearest-even
  return (unsigned short)(u >> 16);
}
__device__ __forceinline__ float bf2f(unsigned short s) {
  return __uint_as_float(((unsigned int)s) << 16);
}
__device__ __forceinline__ float sigm(float x) {
  return 1.0f / (1.0f + __expf(-x));
}
__device__ __forceinline__ float tanh_fast(float x) {
#if __has_builtin(__builtin_amdgcn_tanhf)
  return __builtin_amdgcn_tanhf(x);       // native v_tanh_f32 on gfx1250
#else
  return 2.0f / (1.0f + __expf(-2.0f * x)) - 1.0f;
#endif
}

__device__ __forceinline__ v8f wmma_bf16(v16bf a, v16bf b, v8f c) {
  // D = A(16x32 bf16) x B(32x16 bf16) + C(16x16 f32)
  return __builtin_amdgcn_wmma_f32_16x16x32_bf16(false, a, false, b, (short)0, c, false, false);
}

// A-fragment (16x32 bf16) from LDS h buffer (row-major, stride HP halfs).
// Lanes 0-15: M=lane, K = kt*32 + {0..7, 16..23}; lanes 16-31: M=lane-16, K = +8.
__device__ __forceinline__ v16bf loadA(const unsigned short* hbuf, int mt, int kt, int lane) {
  int m  = mt * 16 + (lane & 15);
  int kb = kt * 32 + ((lane >> 4) << 3);
  const unsigned short* p = hbuf + m * HP + kb;
  BF16Frag r;
  r.q[0] = *(const uint4*)(p);        // k .. k+7
  r.q[1] = *(const uint4*)(p + 16);   // k+16 .. k+23
  return r.v;
}

// B-fragment (32x16 bf16) from pre-packed global weights: fragment (nt,kt) is
// 32 lanes x 16 halfs contiguous -> each lane loads 32B coalesced.
__device__ __forceinline__ v16bf loadB(const unsigned short* wp, int nt, int kt, int lane) {
  const unsigned short* p = wp + ((size_t)((nt << 3) + kt) * 32 + lane) * 16;
  BF16Frag r;
  r.q[0] = *(const uint4*)(p);
  r.q[1] = *(const uint4*)(p + 8);
  return r.v;
}

// Pack W (1024x256 f32 row-major, gates = h @ W.T) into bf16 WMMA B-fragments.
// Fragment f = n0*8 + k0; lane l col n = n0*16 + l%16; half i -> k = k0*32 + 8*(l/16) + (i%8) + 16*(i/8).
__global__ void pack_weights(const float* __restrict__ Whh0,
                             const float* __restrict__ Wih1,
                             const float* __restrict__ Whh1,
                             unsigned short* __restrict__ wsp) {
  int idx = blockIdx.x * 256 + threadIdx.x;      // 0 .. 3*262144-1
  int mat = idx >> 18;
  int p   = idx & 0x3FFFF;
  int f   = p >> 9;
  int rem = p & 511;
  int l   = rem >> 4;
  int i   = rem & 15;
  int n0  = f >> 3;
  int k0  = f & 7;
  int n   = n0 * 16 + (l & 15);
  int k   = k0 * 32 + ((l >> 4) << 3) + (i & 7) + ((i >> 3) << 4);
  const float* W = (mat == 0) ? Whh0 : ((mat == 1) ? Wih1 : Whh1);
  wsp[idx] = f2bf(W[n * HDIM + k]);
}

__global__ void __launch_bounds__(256, 1)
trajectory_lstm(const float* __restrict__ lastpos, const float* __restrict__ hidden,
                const float* __restrict__ cell,    const float* __restrict__ Wih0,
                const float* __restrict__ bih0,    const float* __restrict__ bhh0,
                const float* __restrict__ bih1,    const float* __restrict__ bhh1,
                const float* __restrict__ Wpos,    const float* __restrict__ bpos,
                const float* __restrict__ Wmode,   const float* __restrict__ bmode,
                const unsigned short* __restrict__ wpack,
                float* __restrict__ out) {
  __shared__ __align__(16) unsigned short h0_lds[R * HP];
  __shared__ __align__(16) unsigned short h1_lds[R * HP];
  __shared__ float x_lds[R * 2];
  // Per-gate-column constants staged in LDS to keep their live range short.
  __shared__ float bias0_l[4 * HDIM];
  __shared__ float bias1_l[4 * HDIM];
  __shared__ float wx0_l[4 * HDIM];
  __shared__ float wx1_l[4 * HDIM];
  __shared__ float wpos_lds[6 * HDIM];
  __shared__ float bpos_lds[6];
  __shared__ float wmode_lds[3 * HDIM];
  __shared__ float bmode_lds[3];

  const int tid  = threadIdx.x;
  const int lane = tid & 31;
  const int wv   = tid >> 5;          // wave 0..7 owns hidden units [32w, 32w+32)
  const int b0   = blockIdx.x * R;

  const unsigned short* whh0p = wpack;
  const unsigned short* wih1p = wpack + 262144;
  const unsigned short* whh1p = wpack + 524288;

  // ---- initialize LDS state ----
  for (int idx = tid; idx < R * HDIM; idx += 256) {
    int m = idx >> 8, j = idx & 255;
    h0_lds[m * HP + j] = f2bf(hidden[(size_t)(b0 + m) * HDIM + j]);
    h1_lds[m * HP + j] = f2bf(hidden[(size_t)B_TOTAL * HDIM + (size_t)(b0 + m) * HDIM + j]);
  }
  for (int idx = tid; idx < R * 2; idx += 256) x_lds[idx] = lastpos[(size_t)b0 * 2 + idx];
  for (int idx = tid; idx < 4 * HDIM; idx += 256) {
    bias0_l[idx] = bih0[idx] + bhh0[idx];
    bias1_l[idx] = bih1[idx] + bhh1[idx];
    wx0_l[idx]   = Wih0[idx * 2 + 0];
    wx1_l[idx]   = Wih0[idx * 2 + 1];
  }
  for (int idx = tid; idx < 6 * HDIM; idx += 256) wpos_lds[idx] = Wpos[idx];
  for (int idx = tid; idx < 3 * HDIM; idx += 256) wmode_lds[idx] = Wmode[idx];
  if (tid < 6) bpos_lds[tid] = bpos[tid];
  if (tid < 3) bmode_lds[tid] = bmode[tid];

  // ---- cell state as WMMA C-fragments: c[mt][jt][v] <-> (m = mt*16+v+8*(lane/16), j) ----
  v8f c0[2][2], c1[2][2];
#pragma unroll
  for (int mt = 0; mt < 2; ++mt)
#pragma unroll
    for (int jt = 0; jt < 2; ++jt) {
      int j = wv * 32 + jt * 16 + (lane & 15);
#pragma unroll
      for (int v = 0; v < 8; ++v) {
        int m = b0 + mt * 16 + v + ((lane >> 4) << 3);
        c0[mt][jt][v] = cell[(size_t)m * HDIM + j];
        c1[mt][jt][v] = cell[(size_t)B_TOTAL * HDIM + (size_t)m * HDIM + j];
      }
    }
  __syncthreads();

  const int jbase = wv * 32 + (lane & 15);   // gate-column base for this lane

  for (int ts = 0; ts < T_STEPS; ++ts) {
    v8f acc[2][8];
    // ================= layer 0: gates = x@Wih0.T + h0@Whh0.T + biases =================
    {
      float b0c[8], w0c[8], w1c[8];
#pragma unroll
      for (int fi = 0; fi < 8; ++fi) {
        int jcol = (fi >> 1) * HDIM + (fi & 1) * 16 + jbase;
        b0c[fi] = bias0_l[jcol];
        w0c[fi] = wx0_l[jcol];
        w1c[fi] = wx1_l[jcol];
      }
#pragma unroll
      for (int mt = 0; mt < 2; ++mt)
#pragma unroll
        for (int v = 0; v < 8; ++v) {
          int m = mt * 16 + v + ((lane >> 4) << 3);
          float x0 = x_lds[m * 2 + 0], x1 = x_lds[m * 2 + 1];
#pragma unroll
          for (int fi = 0; fi < 8; ++fi)
            acc[mt][fi][v] = b0c[fi] + x0 * w0c[fi] + x1 * w1c[fi];
        }
    }
#pragma unroll 2
    for (int kt = 0; kt < 8; ++kt) {
      v16bf a0 = loadA(h0_lds, 0, kt, lane);
      v16bf a1 = loadA(h0_lds, 1, kt, lane);
#pragma unroll
      for (int fi = 0; fi < 8; ++fi) {
        int nt = (fi >> 1) * 16 + wv * 2 + (fi & 1);
        v16bf b = loadB(whh0p, nt, kt, lane);
        acc[0][fi] = wmma_bf16(a0, b, acc[0][fi]);
        acc[1][fi] = wmma_bf16(a1, b, acc[1][fi]);
      }
    }
    v8f h0new[2][2];
#pragma unroll
    for (int mt = 0; mt < 2; ++mt)
#pragma unroll
      for (int jt = 0; jt < 2; ++jt)
#pragma unroll
        for (int v = 0; v < 8; ++v) {
          float gi = sigm(acc[mt][0 * 2 + jt][v]);
          float gf = sigm(acc[mt][1 * 2 + jt][v]);
          float gg = tanh_fast(acc[mt][2 * 2 + jt][v]);
          float go = sigm(acc[mt][3 * 2 + jt][v]);
          float cn = gf * c0[mt][jt][v] + gi * gg;
          c0[mt][jt][v] = cn;
          h0new[mt][jt][v] = go * tanh_fast(cn);
        }
    __syncthreads();                    // everyone done reading old h0
#pragma unroll
    for (int mt = 0; mt < 2; ++mt)
#pragma unroll
      for (int jt = 0; jt < 2; ++jt) {
        int j = wv * 32 + jt * 16 + (lane & 15);
#pragma unroll
        for (int v = 0; v < 8; ++v) {
          int m = mt * 16 + v + ((lane >> 4) << 3);
          h0_lds[m * HP + j] = f2bf(h0new[mt][jt][v]);
        }
      }
    __syncthreads();

    // ================= layer 1: gates = h0new@Wih1.T + h1@Whh1.T + biases =================
    {
      float b1c[8];
#pragma unroll
      for (int fi = 0; fi < 8; ++fi)
        b1c[fi] = bias1_l[(fi >> 1) * HDIM + (fi & 1) * 16 + jbase];
#pragma unroll
      for (int mt = 0; mt < 2; ++mt)
#pragma unroll
        for (int fi = 0; fi < 8; ++fi)
#pragma unroll
          for (int v = 0; v < 8; ++v)
            acc[mt][fi][v] = b1c[fi];
    }
#pragma unroll 2
    for (int kt = 0; kt < 8; ++kt) {
      v16bf a0 = loadA(h0_lds, 0, kt, lane);
      v16bf a1 = loadA(h0_lds, 1, kt, lane);
#pragma unroll
      for (int fi = 0; fi < 8; ++fi) {
        int nt = (fi >> 1) * 16 + wv * 2 + (fi & 1);
        v16bf b = loadB(wih1p, nt, kt, lane);
        acc[0][fi] = wmma_bf16(a0, b, acc[0][fi]);
        acc[1][fi] = wmma_bf16(a1, b, acc[1][fi]);
      }
    }
#pragma unroll 2
    for (int kt = 0; kt < 8; ++kt) {
      v16bf a0 = loadA(h1_lds, 0, kt, lane);
      v16bf a1 = loadA(h1_lds, 1, kt, lane);
#pragma unroll
      for (int fi = 0; fi < 8; ++fi) {
        int nt = (fi >> 1) * 16 + wv * 2 + (fi & 1);
        v16bf b = loadB(whh1p, nt, kt, lane);
        acc[0][fi] = wmma_bf16(a0, b, acc[0][fi]);
        acc[1][fi] = wmma_bf16(a1, b, acc[1][fi]);
      }
    }
    v8f h1new[2][2];
#pragma unroll
    for (int mt = 0; mt < 2; ++mt)
#pragma unroll
      for (int jt = 0; jt < 2; ++jt)
#pragma unroll
        for (int v = 0; v < 8; ++v) {
          float gi = sigm(acc[mt][0 * 2 + jt][v]);
          float gf = sigm(acc[mt][1 * 2 + jt][v]);
          float gg = tanh_fast(acc[mt][2 * 2 + jt][v]);
          float go = sigm(acc[mt][3 * 2 + jt][v]);
          float cn = gf * c1[mt][jt][v] + gi * gg;
          c1[mt][jt][v] = cn;
          h1new[mt][jt][v] = go * tanh_fast(cn);
        }
    __syncthreads();                    // everyone done reading old h1
#pragma unroll
    for (int mt = 0; mt < 2; ++mt)
#pragma unroll
      for (int jt = 0; jt < 2; ++jt) {
        int j = wv * 32 + jt * 16 + (lane & 15);
#pragma unroll
        for (int v = 0; v < 8; ++v) {
          int m = mt * 16 + v + ((lane >> 4) << 3);
          h1_lds[m * HP + j] = f2bf(h1new[mt][jt][v]);
        }
      }
    __syncthreads();

    // ================= position head: pos = h1 @ Wpos.T + bpos (6 outputs/row) ==========
    if (tid < R * 6) {
      int r = tid / 6, cc = tid % 6;
      const unsigned short* hr = &h1_lds[r * HP];
      const float* wr = &wpos_lds[cc * HDIM];
      float s = bpos_lds[cc];
      for (int k = 0; k < HDIM; ++k) s += bf2f(hr[k]) * wr[k];
      out[((size_t)(b0 + r) * T_STEPS + ts) * 6 + cc] = s;
      if (cc < 2) x_lds[r * 2 + cc] = s;     // next-step input = pos[:,0,:]
    }
    __syncthreads();
  }

  // ================= mode head: softmax(h1_final @ Wmode.T + bmode) =================
  if (tid < R) {
    int r = tid;
    float vv[3];
    for (int mm = 0; mm < 3; ++mm) {
      float s = bmode_lds[mm];
      const float* wr = &wmode_lds[mm * HDIM];
      for (int k = 0; k < HDIM; ++k) s += bf2f(h1_lds[r * HP + k]) * wr[k];
      vv[mm] = s;
    }
    float mx = fmaxf(vv[0], fmaxf(vv[1], vv[2]));
    float e0 = __expf(vv[0] - mx), e1 = __expf(vv[1] - mx), e2 = __expf(vv[2] - mx);
    float inv = 1.0f / (e0 + e1 + e2);
    size_t base = (size_t)B_TOTAL * T_STEPS * 6;
    out[base + (size_t)(b0 + r) * 3 + 0] = e0 * inv;
    out[base + (size_t)(b0 + r) * 3 + 1] = e1 * inv;
    out[base + (size_t)(b0 + r) * 3 + 2] = e2 * inv;
  }
}

extern "C" void kernel_launch(void* const* d_in, const int* in_sizes, int n_in,
                              void* d_out, int out_size, void* d_ws, size_t ws_size,
                              hipStream_t stream) {
  const float* lastpos = (const float*)d_in[0];
  const float* hidden  = (const float*)d_in[1];
  const float* cell    = (const float*)d_in[2];
  const float* Wih0    = (const float*)d_in[3];
  const float* Whh0    = (const float*)d_in[4];
  const float* bih0    = (const float*)d_in[5];
  const float* bhh0    = (const float*)d_in[6];
  const float* Wih1    = (const float*)d_in[7];
  const float* Whh1    = (const float*)d_in[8];
  const float* bih1    = (const float*)d_in[9];
  const float* bhh1    = (const float*)d_in[10];
  const float* Wpos    = (const float*)d_in[11];
  const float* bpos    = (const float*)d_in[12];
  const float* Wmode   = (const float*)d_in[13];
  const float* bmode   = (const float*)d_in[14];
  unsigned short* wpack = (unsigned short*)d_ws;   // 3 * 262144 bf16 = 1.5 MB
  float* out = (float*)d_out;

  pack_weights<<<(3 * 262144) / 256, 256, 0, stream>>>(Whh0, Wih1, Whh1, wpack);
  trajectory_lstm<<<B_TOTAL / R, 256, 0, stream>>>(
      lastpos, hidden, cell, Wih0, bih0, bhh0, bih1, bhh1,
      Wpos, bpos, Wmode, bmode, wpack, out);
}
